// VectorQuantizer_19404662243549
// MI455X (gfx1250) — matrix operand order repaired and checked by reading, not compiled, not run on hardware
//
#include <hip/hip_runtime.h>

// Problem constants (from reference)
#define BATCH   16
#define CDIM    256     // channels == embedding dim D
#define HWDIM   4096    // H*W = 64*64
#define KCODES  1024    // codebook entries
#define TOK_WG  128     // tokens per workgroup (4 waves x 32 tokens)
#define LDA     264     // LDS row pitch in bf16 elements (256 + 8 pad)

typedef __attribute__((ext_vector_type(16))) __bf16 v16bf;
typedef __attribute__((ext_vector_type(8)))  float  v8f;

union Frag16 { uint4 q[2]; v16bf v; };

#if __has_builtin(__builtin_amdgcn_global_load_async_to_lds_b128) && \
    __has_builtin(__builtin_amdgcn_s_wait_asynccnt)
#define VQ_USE_ASYNC 1
#else
#define VQ_USE_ASYNC 0
#endif

// Builtin signature (from hipcc diagnostic): takes pointers to 128-bit int
// vectors, global (AS1) source and LDS (AS3) destination, plus 2 imm ints.
typedef int v4i __attribute__((vector_size(16)));
typedef __attribute__((address_space(1))) v4i glb_v4i;
typedef __attribute__((address_space(3))) v4i lds_v4i;

__device__ __forceinline__ unsigned short f2bf(float f) {
    unsigned u = __float_as_uint(f);
    // round-to-nearest-even truncate to bf16
    unsigned r = (u + 0x7FFFu + ((u >> 16) & 1u)) >> 16;
    return (unsigned short)r;
}

// ---------------------------------------------------------------------------
// Kernel 1: embedding fp32 -> bf16 copy in d_ws, plus ||e_k||^2 per code.
// ---------------------------------------------------------------------------
__global__ void vq_prep_kernel(const float* __restrict__ emb,
                               unsigned short* __restrict__ ebf,
                               float* __restrict__ norms) {
    int k = blockIdx.x;
    int t = threadIdx.x;
    float v = emb[(size_t)k * CDIM + t];
    ebf[(size_t)k * CDIM + t] = f2bf(v);

    float sq = v * v;
    #pragma unroll
    for (int m = 16; m >= 1; m >>= 1)
        sq += __shfl_xor(sq, m, 32);

    __shared__ float partial[8];
    int wid = t >> 5, lane = t & 31;
    if (lane == 0) partial[wid] = sq;
    __syncthreads();
    if (t == 0) {
        float s = 0.f;
        #pragma unroll
        for (int i = 0; i < 8; ++i) s += partial[i];
        norms[k] = s;
    }
}

// ---------------------------------------------------------------------------
// Kernel 2: per-token argmin over codes via bf16 WMMA (2 token tiles / wave,
// double-buffered async B staging), then gather fp32 rows.
// Block: 128 threads (4 waves). Grid: N tokens / 128.
// ---------------------------------------------------------------------------
__global__ void __launch_bounds__(128, 1)
vq_main_kernel(const float* __restrict__ in,          // [B, C, H*W] fp32
               const float* __restrict__ emb,         // [K, D] fp32
               const unsigned short* __restrict__ ebf,// [K, D] bf16
               const float* __restrict__ norms,       // [K] fp32
               float* __restrict__ out) {             // [B, C, H*W] fp32
    __shared__ __align__(16) unsigned short Alds[TOK_WG][LDA];   // token-major
    __shared__ __align__(16) unsigned short Blds[2][16][LDA];    // double buffer
    __shared__ int idx_lds[TOK_WG];

    const int tid = threadIdx.x;
    const long T0 = (long)blockIdx.x * TOK_WG;       // first flat token (b,h,w)
    const int  b    = (int)(T0 >> 12);               // /4096
    const int  pos0 = (int)(T0 & 4095);
    const float* inb = in + (size_t)b * CDIM * HWDIM + pos0;

    // ---- Stage A tile: 128 tokens x 256 channels, fp32 -> bf16, transposed.
    {
        const int pos = tid;                          // 0..127
        #pragma unroll 4
        for (int c2 = 0; c2 < 128; ++c2) {
            int c = c2 * 2;
            float f0 = inb[(size_t)c * HWDIM + pos];
            float f1 = inb[(size_t)(c + 1) * HWDIM + pos];
            unsigned packed = (unsigned)f2bf(f0) | ((unsigned)f2bf(f1) << 16);
            *(unsigned*)&Alds[pos][c] = packed;
        }
    }
    __syncthreads();

    const int lane  = tid & 31;
    const int wid   = tid >> 5;
    const int m0    = wid * 32;       // wave's 32-token tile
    const int lhalf = lane >> 4;      // lane 0-15 vs 16-31
    const int lmod  = lane & 15;

    // ---- A fragments for both 16-token tiles, register-resident.
    // ISA 16-bit A 16x32 layout: lanes 0-15 (row M=lane) hold K {0..7,16..23};
    // lanes 16-31 hold K {8..15,24..31} -> two b128 LDS loads per fragment.
    Frag16 a0[8], a1[8];
    {
        const unsigned short* ar0 = &Alds[m0 + lmod][0];
        const unsigned short* ar1 = &Alds[m0 + 16 + lmod][0];
        #pragma unroll
        for (int kc = 0; kc < 8; ++kc) {
            int kb = kc * 32 + lhalf * 8;
            a0[kc].q[0] = *(const uint4*)(ar0 + kb);
            a0[kc].q[1] = *(const uint4*)(ar0 + kb + 16);
            a1[kc].q[0] = *(const uint4*)(ar1 + kb);
            a1[kc].q[1] = *(const uint4*)(ar1 + kb + 16);
        }
    }

    // ---- B-tile staging helper state: each thread moves 64B (4 x b128).
    const int scode = tid >> 3;              // 0..15
    const int skoff = (tid & 7) * 32;        // 0..224 (bf16 elements)

    // Prefetch chunk 0 into buffer 0.
    {
        const unsigned short* src = ebf + (size_t)scode * CDIM + skoff;
        unsigned short*       dst = &Blds[0][scode][skoff];
#if VQ_USE_ASYNC
        #pragma unroll
        for (int j = 0; j < 4; ++j)
            __builtin_amdgcn_global_load_async_to_lds_b128(
                (glb_v4i*)(src + j * 8), (lds_v4i*)(dst + j * 8), 0, 0);
#else
        #pragma unroll
        for (int j = 0; j < 4; ++j)
            ((uint4*)dst)[j] = ((const uint4*)src)[j];
#endif
    }

    float minval[16];
    int   minidx[16];
    #pragma unroll
    for (int r = 0; r < 16; ++r) { minval[r] = 3.4e38f; minidx[r] = 0; }

    // ---- Main loop: 64 chunks of 16 codes, single barrier per iteration.
    for (int i = 0; i < KCODES / 16; ++i) {
        const int n0 = i * 16;
#if VQ_USE_ASYNC
        __builtin_amdgcn_s_wait_asynccnt(0);   // this wave's chunk-i writes done
#endif
        __syncthreads();                        // all waves: chunk i visible,
                                                // prev compute reads retired
        if (i + 1 < KCODES / 16) {              // prefetch chunk i+1
            const unsigned short* src =
                ebf + (size_t)(n0 + 16 + scode) * CDIM + skoff;
            unsigned short* dst = &Blds[(i + 1) & 1][scode][skoff];
#if VQ_USE_ASYNC
            #pragma unroll
            for (int j = 0; j < 4; ++j)
                __builtin_amdgcn_global_load_async_to_lds_b128(
                    (glb_v4i*)(src + j * 8), (lds_v4i*)(dst + j * 8), 0, 0);
#else
            #pragma unroll
            for (int j = 0; j < 4; ++j)
                ((uint4*)dst)[j] = ((const uint4*)src)[j];
#endif
        }

        float nrm = norms[n0 + lmod];

        v8f c0 = {0.f, 0.f, 0.f, 0.f, 0.f, 0.f, 0.f, 0.f};
        v8f c1 = {0.f, 0.f, 0.f, 0.f, 0.f, 0.f, 0.f, 0.f};
        const unsigned short* brow = &Blds[i & 1][lmod][0];
        // B 32x16 layout: lane holds column n=lane%16; lanes 0-15 cover K {0..7,16..23},
        // lanes 16-31 cover K {8..15,24..31} of each 32-K block (matches A).
        #pragma unroll
        for (int kc = 0; kc < 8; ++kc) {
            Frag16 bf;
            int kb = kc * 32 + lhalf * 8;
            bf.q[0] = *(const uint4*)(brow + kb);
            bf.q[1] = *(const uint4*)(brow + kb + 16);
            c0 = __builtin_amdgcn_wmma_f32_16x16x32_bf16(
                    false, a0[kc].v, false, bf.v, (short)0, c0, false, false);
            c1 = __builtin_amdgcn_wmma_f32_16x16x32_bf16(
                    false, a1[kc].v, false, bf.v, (short)0, c1, false, false);
        }

        // dist = ||e||^2 - 2 * dot  (token norm constant -> argmin-invariant)
        const int nidx = n0 + lmod;
        #pragma unroll
        for (int r = 0; r < 8; ++r) {
            float d0 = nrm - 2.0f * c0[r];
            float d1 = nrm - 2.0f * c1[r];
            if (d0 < minval[r])     { minval[r] = d0;     minidx[r] = nidx; }
            if (d1 < minval[8 + r]) { minval[8 + r] = d1; minidx[8 + r] = nidx; }
        }
    }

    // ---- Cross-lane argmin: row m = r + 8*lhalf lives in 16 lanes of a half.
    #pragma unroll
    for (int r = 0; r < 16; ++r) {
        float v = minval[r];
        int  id = minidx[r];
        #pragma unroll
        for (int m = 1; m < 16; m <<= 1) {
            float ov = __shfl_xor(v, m, 32);
            int   oi = __shfl_xor(id, m, 32);
            if (ov < v || (ov == v && oi < id)) { v = ov; id = oi; }
        }
        if (lmod == 0)
            idx_lds[m0 + (r >> 3) * 16 + (r & 7) + 8 * lhalf] = id;
    }
    __syncthreads();

    // ---- Output: quantized = emb[idx] written back NCHW, lanes along W.
    {
        float* outb = out + (size_t)b * CDIM * HWDIM + pos0;
        const int t = tid;                          // one token per thread
        const int myidx = idx_lds[t];
        const float4* erow4 = (const float4*)(emb + (size_t)myidx * CDIM);
        #pragma unroll 4
        for (int c4 = 0; c4 < 64; ++c4) {
            float4 v = erow4[c4];
            outb[(size_t)(c4 * 4 + 0) * HWDIM + t] = v.x;
            outb[(size_t)(c4 * 4 + 1) * HWDIM + t] = v.y;
            outb[(size_t)(c4 * 4 + 2) * HWDIM + t] = v.z;
            outb[(size_t)(c4 * 4 + 3) * HWDIM + t] = v.w;
        }
    }
}

// ---------------------------------------------------------------------------
extern "C" void kernel_launch(void* const* d_in, const int* in_sizes, int n_in,
                              void* d_out, int out_size, void* d_ws, size_t ws_size,
                              hipStream_t stream) {
    const float* inputs = (const float*)d_in[0];   // [16,256,64,64] fp32
    const float* emb    = (const float*)d_in[1];   // [1024,256] fp32
    float* out          = (float*)d_out;           // [16,256,64,64] fp32

    unsigned short* ebf = (unsigned short*)d_ws;                       // 512 KB bf16
    float* norms        = (float*)((char*)d_ws + (size_t)KCODES * CDIM * 2);

    vq_prep_kernel<<<KCODES, 256, 0, stream>>>(emb, ebf, norms);

    const int n_tokens = BATCH * HWDIM;            // 65536
    vq_main_kernel<<<n_tokens / TOK_WG, 128, 0, stream>>>(inputs, emb, ebf, norms, out);
}
